// VectorQuantization_14809047237049
// MI455X (gfx1250) — matrix operand order repaired and checked by reading, compiled, not run on hardware
//
#include <hip/hip_runtime.h>

typedef __attribute__((ext_vector_type(16))) _Float16 v16h;
typedef __attribute__((ext_vector_type(8)))  _Float16 v8h;
typedef __attribute__((ext_vector_type(8)))  float    v8f;
typedef __attribute__((ext_vector_type(4)))  float    v4f;
typedef __attribute__((ext_vector_type(4)))  int      v4i;
typedef __attribute__((ext_vector_type(2)))  float    v2f;

// Problem geometry (fixed by the reference)
static constexpr int Bn  = 32;
static constexpr int Cd  = 64;      // embedding dim
static constexpr int HW  = 64 * 64; // 4096
static constexpr int Np  = Bn * HW; // 131072 points
static constexpr int Kc  = 1024;    // codebook size
static constexpr float BETA = 0.25f;

// ---------------- Kernel 0: emb -> f16, e_sq (repacked per-lane) ----------------
// esqP[(n&15)*64 + (n>>4)] = ||emb[n]||^2  -> each lane's 64 bias values contiguous
__global__ void k_prep_emb(const float* __restrict__ emb,
                           _Float16* __restrict__ emb16,
                           float* __restrict__ esqP) {
    int t = blockIdx.x * blockDim.x + threadIdx.x;   // 0..65535
    if (t < Kc * Cd) emb16[t] = (_Float16)emb[t];
    if (t < Kc) {
        const float* r = emb + t * Cd;
        float s = 0.f;
        #pragma unroll 8
        for (int i = 0; i < Cd; ++i) s += r[i] * r[i];
        esqP[(t & 15) * 64 + (t >> 4)] = s;
    }
}

// ---------------- Kernel 1: x [b,c,h,w] -> xt [n=b*hw, c] f16 ----------------
__global__ __launch_bounds__(256) void k_transpose(const float* __restrict__ x,
                                                   _Float16* __restrict__ xt) {
    __shared__ float tile[64 * 65];                  // 65 pad -> conflict-free
    int blk = blockIdx.x;                            // 0..2047
    int b   = blk >> 6;
    int hw0 = (blk & 63) << 6;
    int t   = threadIdx.x;
    #pragma unroll
    for (int k = 0; k < 16; ++k) {
        int idx = t + (k << 8);                      // 0..4095
        int c = idx >> 6, j = idx & 63;
        tile[c * 65 + j] = x[((size_t)(b * 64 + c) << 12) + hw0 + j];
    }
    __syncthreads();
    #pragma unroll
    for (int k = 0; k < 16; ++k) {
        int idx = t + (k << 8);
        int j = idx >> 6, c = idx & 63;
        xt[((size_t)((b << 12) + hw0 + j) << 6) + c] = (_Float16)tile[c * 65 + j];
    }
}

// ---------------- Kernel 2: fused GEMM + argmin (WMMA f16) ----------------
// Wave handles 16 rows of X; loops 32 codes/iter (two 16-code tiles with
// independent accumulators so compare/select of one tile hides the
// WMMA->VALU hazard of the other).
// score = x.e - 0.5*||e||^2 ; argmax(score) == argmin(l2)
__global__ __launch_bounds__(256) void k_argmin(const _Float16* __restrict__ xt,
                                                const _Float16* __restrict__ emb16,
                                                const float* __restrict__ esqP,
                                                int* __restrict__ bestidx) {
    int lane = threadIdx.x & 31;
    int wave = threadIdx.x >> 5;
    int r0   = (blockIdx.x * 8 + wave) * 16;         // row-tile origin
    int m16  = lane & 15;
    int kh   = lane >> 4;

    // A operand: 16x32 f16 per chunk; lane half 0 -> K[0..15];
    // lane half 1 -> K[16..31] (matches B's contiguous order). Two chunks cover K=64.
    const _Float16* arow = xt + ((size_t)(r0 + m16) << 6);
    union { v16h v; v8h h[2]; } a0u, a1u;
    a0u.h[0] = *(const v8h*)(arow + kh * 16);
    a0u.h[1] = *(const v8h*)(arow + kh * 16 + 8);
    a1u.h[0] = *(const v8h*)(arow + 32 + kh * 16);
    a1u.h[1] = *(const v8h*)(arow + 32 + kh * 16 + 8);

    const float* esqp = esqP + m16 * 64;             // this lane's bias stream

    float best[8];
    int   bidx[8];
    #pragma unroll
    for (int m = 0; m < 8; ++m) { best[m] = -3.402823e38f; bidx[m] = 0; }

    for (int t = 0; t < 64; t += 2) {
        int n0 = t << 4;
        // B operand: 32x16 f16; lane holds 16 contiguous K-halfs of emb row n
        const _Float16* browA = emb16 + ((size_t)(n0 + m16) << 6) + kh * 16;
        v16h bA0 = *(const v16h*)(browA);
        v16h bA1 = *(const v16h*)(browA + 32);
        const _Float16* browB = browA + (16 << 6);   // codes n0+16..n0+31
        v16h bB0 = *(const v16h*)(browB);
        v16h bB1 = *(const v16h*)(browB + 32);

        v2f nh2 = *(const v2f*)(esqp + t);           // two tiles' biases at once
        float nhA = -0.5f * nh2.x;
        float nhB = -0.5f * nh2.y;

        v8f cA, cB;
        #pragma unroll
        for (int m = 0; m < 8; ++m) cA[m] = nhA;     // fold -0.5||e||^2 into C
        cA = __builtin_amdgcn_wmma_f32_16x16x32_f16(false, a0u.v, false, bA0,
                                                    (short)0, cA, false, false);
        cA = __builtin_amdgcn_wmma_f32_16x16x32_f16(false, a1u.v, false, bA1,
                                                    (short)0, cA, false, false);
        #pragma unroll
        for (int m = 0; m < 8; ++m) cB[m] = nhB;
        cB = __builtin_amdgcn_wmma_f32_16x16x32_f16(false, a0u.v, false, bB0,
                                                    (short)0, cB, false, false);
        cB = __builtin_amdgcn_wmma_f32_16x16x32_f16(false, a1u.v, false, bB1,
                                                    (short)0, cB, false, false);

        int nA = n0 + m16;
        #pragma unroll
        for (int m = 0; m < 8; ++m) {                // strict > keeps first occurrence
            if (cA[m] > best[m]) { best[m] = cA[m]; bidx[m] = nA; }
        }
        int nB = nA + 16;
        #pragma unroll
        for (int m = 0; m < 8; ++m) {
            if (cB[m] > best[m]) { best[m] = cB[m]; bidx[m] = nB; }
        }
    }

    // Reduce argmax across the 16 lanes of each N-group (smaller index on ties)
    #pragma unroll
    for (int off = 8; off >= 1; off >>= 1) {
        #pragma unroll
        for (int m = 0; m < 8; ++m) {
            float s2 = __shfl_xor(best[m], off, 16);
            int   i2 = __shfl_xor(bidx[m], off, 16);
            if (s2 > best[m] || (s2 == best[m] && i2 < bidx[m])) {
                best[m] = s2; bidx[m] = i2;
            }
        }
    }
    if (m16 == 0) {
        int mb = kh * 8;                             // lanes 16-31 hold rows M=8..15
        #pragma unroll
        for (int m = 0; m < 8; ++m) bestidx[r0 + mb + m] = bidx[m];
    }
}

// ---------------- Kernel 3: gather emb[idx] -> out (b,c,h,w) + loss partials ----------------
// 4 elements per thread: v4i idx load, v4f x load (NT), v4f out store (NT)
__global__ __launch_bounds__(256) void k_gather(const float* __restrict__ x,
                                                const float* __restrict__ emb,
                                                const int* __restrict__ bestidx,
                                                float* __restrict__ out,
                                                float* __restrict__ partial) {
    __shared__ float wsum[8];
    size_t o = ((size_t)blockIdx.x * 256 + threadIdx.x) * 4;
    int hw = (int)(o & 4095);
    int c  = (int)((o >> 12) & 63);
    int b  = (int)(o >> 18);
    int n  = (b << 12) + hw;                         // multiple of 4 -> v4i aligned

    v4i kk = *(const v4i*)(bestidx + n);
    v4f xv = __builtin_nontemporal_load((const v4f*)(x + o));
    v4f q;
    q.x = emb[((size_t)kk.x << 6) + c];              // full-precision gather (L2-resident)
    q.y = emb[((size_t)kk.y << 6) + c];
    q.z = emb[((size_t)kk.z << 6) + c];
    q.w = emb[((size_t)kk.w << 6) + c];
    __builtin_nontemporal_store(q, (v4f*)(out + o));

    v4f d = q - xv;
    float s = d.x * d.x + d.y * d.y + d.z * d.z + d.w * d.w;
    #pragma unroll
    for (int off = 16; off >= 1; off >>= 1) s += __shfl_xor(s, off, 32);
    int lane = threadIdx.x & 31, wave = threadIdx.x >> 5;
    if (lane == 0) wsum[wave] = s;
    __syncthreads();
    if (threadIdx.x == 0) {
        float tsum = 0.f;
        #pragma unroll
        for (int i = 0; i < 8; ++i) tsum += wsum[i];
        partial[blockIdx.x] = tsum;
    }
}

// ---------------- Kernel 4: deterministic final loss reduce ----------------
__global__ void k_loss(const float* __restrict__ partial, float* __restrict__ lossOut) {
    __shared__ float sh[256];
    float s = 0.f;
    for (int i = threadIdx.x; i < 8192; i += 256) s += partial[i];
    sh[threadIdx.x] = s;
    __syncthreads();
    for (int off = 128; off >= 1; off >>= 1) {
        if ((int)threadIdx.x < off) sh[threadIdx.x] += sh[threadIdx.x + off];
        __syncthreads();
    }
    if (threadIdx.x == 0)
        *lossOut = (1.0f + BETA) * sh[0] / (float)((size_t)Np * Cd);
}

extern "C" void kernel_launch(void* const* d_in, const int* in_sizes, int n_in,
                              void* d_out, int out_size, void* d_ws, size_t ws_size,
                              hipStream_t stream) {
    const float* x   = (const float*)d_in[0];   // [32,64,64,64]
    const float* emb = (const float*)d_in[1];   // [1024,64]
    float* out = (float*)d_out;                 // 8388608 quantized + 1 loss

    char* ws = (char*)d_ws;
    size_t off = 0;
    _Float16* xt    = (_Float16*)(ws + off); off += (size_t)Np * Cd * 2;   // 16 MiB
    _Float16* emb16 = (_Float16*)(ws + off); off += (size_t)Kc * Cd * 2;   // 128 KiB
    float*    esqP  = (float*)   (ws + off); off += (size_t)Kc * 4;        // 4 KiB
    int*      bidx  = (int*)     (ws + off); off += (size_t)Np * 4;        // 512 KiB
    float*    part  = (float*)   (ws + off); off += 8192 * 4;              // 32 KiB

    k_prep_emb <<<256,  256, 0, stream>>>(emb, emb16, esqP);
    k_transpose<<<2048, 256, 0, stream>>>(x, xt);
    k_argmin   <<<1024, 256, 0, stream>>>(xt, emb16, esqP, bidx);
    k_gather   <<<8192, 256, 0, stream>>>(x, emb, bidx, out, part);
    k_loss     <<<1,    256, 0, stream>>>(part, out + (size_t)Np * Cd);
}
